// SwinTransformerBlock_85822036509210
// MI455X (gfx1250) — matrix operand order repaired and checked by reading, not compiled, not run on hardware
//
#include <hip/hip_runtime.h>
#include <hip/hip_bf16.h>

// ---------------- problem constants ----------------
#define BD     4
#define HDIM   256
#define WDIM   256
#define CDIM   192
#define NHEAD  6
#define HD     32
#define WSZ    8
#define SHIFT  4
#define NTOK   64          // tokens per window
#define NWIMG  1024        // windows per image
#define NWTOT  4096        // total windows
#define HIDDEN 768
#define QSCALE 0.17677669529663687f   // 32^-0.5

typedef __attribute__((ext_vector_type(8)))  unsigned short u16x8;
typedef __attribute__((ext_vector_type(16))) unsigned short u16x16;
typedef __attribute__((ext_vector_type(16))) __bf16         bf16x16;
typedef __attribute__((ext_vector_type(8)))  float          f32x8;

// ---------------- scalar helpers ----------------
__device__ inline unsigned short f2bf(float f) {
    unsigned int u = __builtin_bit_cast(unsigned int, f);
    unsigned int r = u + 0x7fffu + ((u >> 16) & 1u);   // RNE
    return (unsigned short)(r >> 16);
}

// Async copy of one 16-byte chunk: global (saddr + voffset) -> LDS (lds_off).
// GLOBAL_LOAD_ASYNC_TO_LDS_B128, GVS mode; tracked by ASYNCcnt.
__device__ inline void async_copy_b128(unsigned lds_off, unsigned goff, const void* sbase) {
    asm volatile("global_load_async_to_lds_b128 %0, %1, %2"
                 :: "v"(lds_off), "v"(goff), "s"(sbase) : "memory");
}
__device__ inline void wait_async0() {
    asm volatile("s_wait_asynccnt 0" ::: "memory");
}

// ---------------- WMMA fragment loaders (CDNA5 bf16 16x16x32 layouts) ----------------
// A (MxK, 16x32): lane L -> row L%16; lanes<16: K k0..k0+7 (v0-3) and k0+16..k0+23 (v4-7);
//                 lanes>=16: K k0+8.. and k0+24..
__device__ inline bf16x16 load_a_frag(const unsigned short* base, int row, int k0, int stride) {
    int lane = threadIdx.x & 31;
    int r  = row + (lane & 15);
    int kh = k0 + ((lane >> 4) << 3);
    const u16x8* p0 = (const u16x8*)(base + (size_t)r * stride + kh);
    const u16x8* p1 = (const u16x8*)(base + (size_t)r * stride + kh + 16);
    union { u16x16 v; u16x8 h[2]; } u;
    u.h[0] = *p0; u.h[1] = *p1;
    return __builtin_bit_cast(bf16x16, u.v);
}
// B (KxN, 32x16) stored as rows over N with K contiguous (i.e. src[n][k]):
// lane L -> col L%16; lanes<16 hold K k0..k0+7 and k0+16..k0+23, lanes>=16 hold K k0+8.. and k0+24..
__device__ inline bf16x16 load_b_frag(const unsigned short* base, int col, int k0, int stride) {
    int lane = threadIdx.x & 31;
    int n  = col + (lane & 15);
    int kh = k0 + ((lane >> 4) << 3);
    const u16x8* p0 = (const u16x8*)(base + (size_t)n * stride + kh);
    const u16x8* p1 = (const u16x8*)(base + (size_t)n * stride + kh + 16);
    union { u16x16 v; u16x8 h[2]; } u;
    u.h[0] = *p0; u.h[1] = *p1;
    return __builtin_bit_cast(bf16x16, u.v);
}
__device__ inline f32x8 wmma_bf16(bf16x16 a, bf16x16 b, f32x8 c) {
    return __builtin_amdgcn_wmma_f32_16x16x32_bf16(false, a, false, b, (short)0, c, false, false);
}

#define LSTRIDE 200   // LDS row stride in halfwords (multiple of 8 -> 16B aligned rows)

// ---------------- K1: QKV projection (gather shifted windows) ----------------
// grid (4096 windows, 9 col-tiles), 256 threads. out layout per window (bf16):
//   q,k : [sel][head][token][hd]   v : [2][head][hd][token]  (block = 3*6*2048 halfs)
__global__ __launch_bounds__(256) void k_qkv(const float* __restrict__ x,
                                             const float* __restrict__ qkv_w,
                                             const float* __restrict__ qkv_b,
                                             unsigned short* __restrict__ qkv_ws) {
    __shared__ alignas(16) unsigned short lA[64 * LSTRIDE];
    __shared__ alignas(16) unsigned short lB[64 * LSTRIDE];
    int widx = blockIdx.x;
    int n0   = blockIdx.y * 64;
    int b = widx >> 10, rr_ = widx & 1023, hb = rr_ >> 5, wb = rr_ & 31;

    for (int idx = threadIdx.x; idx < 64 * CDIM; idx += 256) {
        int t = idx / CDIM, c = idx % CDIM;
        int i = t >> 3, j = t & 7;
        int hs = ((hb * 8 + i) + SHIFT) & 255;
        int ws = ((wb * 8 + j) + SHIFT) & 255;
        lA[t * LSTRIDE + c] = f2bf(x[(((size_t)b * 65536) + hs * 256 + ws) * CDIM + c]);
    }
    for (int idx = threadIdx.x; idx < 64 * CDIM; idx += 256) {
        int n = idx / CDIM, c = idx % CDIM;
        lB[n * LSTRIDE + c] = f2bf(qkv_w[(size_t)(n0 + n) * CDIM + c]);
    }
    __syncthreads();

    int wave = threadIdx.x >> 5, lane = threadIdx.x & 31;
    for (int t = 0; t < 2; ++t) {
        int tile = wave * 2 + t, mt = tile >> 2, nt = tile & 3;
        f32x8 acc = {0.f,0.f,0.f,0.f,0.f,0.f,0.f,0.f};
        #pragma unroll
        for (int kc = 0; kc < CDIM; kc += 32)
            acc = wmma_bf16(load_a_frag(lA, mt * 16, kc, LSTRIDE),
                            load_b_frag(lB, nt * 16, kc, LSTRIDE), acc);
        int n   = n0 + nt * 16 + (lane & 15);
        int sel = n / CDIM, hh = (n % CDIM) / HD, d = n & 31;
        float bias = qkv_b[n];
        float scl  = (sel == 0) ? QSCALE : 1.0f;
        size_t base = (((size_t)widx * 3 + sel) * NHEAD + hh) * 2048;
        #pragma unroll
        for (int r = 0; r < 8; ++r) {
            int m = mt * 16 + r + ((lane >> 4) << 3);
            float v = (acc[r] + bias) * scl;
            if (sel < 2) qkv_ws[base + m * 32 + d] = f2bf(v);   // token-major
            else         qkv_ws[base + d * 64 + m] = f2bf(v);   // transposed V
        }
    }
}

// ---------------- K2: windowed attention ----------------
// grid (4096, 6 heads), 256 threads
__global__ __launch_bounds__(256) void k_attn(const unsigned short* __restrict__ qkv_ws,
                                              const float* __restrict__ rpb_table,
                                              const int*   __restrict__ rel_idx,
                                              const float* __restrict__ mask_matrix,
                                              unsigned short* __restrict__ attn_out) {
    __shared__ alignas(16) float          Sl[64 * 68];
    __shared__ alignas(16) unsigned short Pl[64 * 72];
    int widx = blockIdx.x, h = blockIdx.y;
    const unsigned short* qb = qkv_ws + (((size_t)widx * 3 + 0) * NHEAD + h) * 2048;
    const unsigned short* kb = qkv_ws + (((size_t)widx * 3 + 1) * NHEAD + h) * 2048;
    const unsigned short* vb = qkv_ws + (((size_t)widx * 3 + 2) * NHEAD + h) * 2048;
    int wave = threadIdx.x >> 5, lane = threadIdx.x & 31;

    // S = q @ k^T   (64x64, K=32)
    for (int t = 0; t < 2; ++t) {
        int tile = wave * 2 + t, mt = tile >> 2, nt = tile & 3;
        f32x8 acc = {0.f,0.f,0.f,0.f,0.f,0.f,0.f,0.f};
        acc = wmma_bf16(load_a_frag(qb, mt * 16, 0, 32),
                        load_b_frag(kb, nt * 16, 0, 32), acc);
        int n = nt * 16 + (lane & 15);
        #pragma unroll
        for (int r = 0; r < 8; ++r) {
            int m = mt * 16 + r + ((lane >> 4) << 3);
            Sl[m * 68 + n] = acc[r];
        }
    }
    __syncthreads();

    // per-row softmax with rel-pos bias + shift mask
    if (threadIdx.x < 64) {
        int i = threadIdx.x;
        const float* mrow = mask_matrix + (size_t)(widx & 1023) * 4096 + i * 64;
        const int*   ridx = rel_idx + i * 64;
        float sv[64];
        float mx = -1e30f;
        for (int j = 0; j < 64; ++j) {
            float s = Sl[i * 68 + j] + rpb_table[ridx[j] * NHEAD + h] + mrow[j];
            sv[j] = s;
            mx = fmaxf(mx, s);
        }
        float sum = 0.f;
        for (int j = 0; j < 64; ++j) { float e = __expf(sv[j] - mx); sv[j] = e; sum += e; }
        float inv = 1.0f / sum;
        for (int j = 0; j < 64; ++j) Pl[i * 72 + j] = f2bf(sv[j] * inv);
    }
    __syncthreads();

    // O = P @ V   (64x32, K=64), one 16x16 tile per wave
    {
        int mt = wave >> 1, nt = wave & 1;
        f32x8 acc = {0.f,0.f,0.f,0.f,0.f,0.f,0.f,0.f};
        #pragma unroll
        for (int kc = 0; kc < 64; kc += 32)
            acc = wmma_bf16(load_a_frag(Pl, mt * 16, kc, 72),
                            load_b_frag(vb, nt * 16, kc, 64), acc);
        int d = nt * 16 + (lane & 15);
        int cc = h * HD + d;
        #pragma unroll
        for (int r = 0; r < 8; ++r) {
            int i = mt * 16 + r + ((lane >> 4) << 3);
            attn_out[((size_t)widx * 64 + i) * CDIM + cc] = f2bf(acc[r]);
        }
    }
}

// ---------------- K3: output projection + window-reverse + un-shift ----------------
__global__ __launch_bounds__(256) void k_proj(const unsigned short* __restrict__ attn_out,
                                              const float* __restrict__ proj_w,
                                              const float* __restrict__ proj_b,
                                              float* __restrict__ xr) {
    __shared__ alignas(16) unsigned short lA[64 * LSTRIDE];
    __shared__ alignas(16) unsigned short lB[64 * LSTRIDE];
    int widx = blockIdx.x, n0 = blockIdx.y * 64;

    // A tile is a pure bf16 copy -> async load-to-LDS (no VGPR round trip).
    {
        const unsigned short* gbase = attn_out + (size_t)widx * 64 * CDIM;
        #pragma unroll
        for (int it = 0; it < 6; ++it) {
            int chunk = threadIdx.x + 256 * it;        // 1536 x 16B chunks
            int row = (chunk * 8) / CDIM;
            int col = (chunk * 8) % CDIM;
            unsigned lds_off = (unsigned)(uintptr_t)&lA[row * LSTRIDE + col];
            unsigned goff    = (unsigned)((row * CDIM + col) * 2);
            async_copy_b128(lds_off, goff, gbase);
        }
    }
    for (int idx = threadIdx.x; idx < 64 * CDIM; idx += 256) {
        int n = idx / CDIM, c = idx % CDIM;
        lB[n * LSTRIDE + c] = f2bf(proj_w[(size_t)(n0 + n) * CDIM + c]);
    }
    wait_async0();
    __syncthreads();

    int wave = threadIdx.x >> 5, lane = threadIdx.x & 31;
    int b = widx >> 10, rr_ = widx & 1023, hb = rr_ >> 5, wb = rr_ & 31;
    for (int t = 0; t < 2; ++t) {
        int tile = wave * 2 + t, mt = tile >> 2, nt = tile & 3;
        f32x8 acc = {0.f,0.f,0.f,0.f,0.f,0.f,0.f,0.f};
        #pragma unroll
        for (int kc = 0; kc < CDIM; kc += 32)
            acc = wmma_bf16(load_a_frag(lA, mt * 16, kc, LSTRIDE),
                            load_b_frag(lB, nt * 16, kc, LSTRIDE), acc);
        int n = n0 + nt * 16 + (lane & 15);
        float bias = proj_b[n];
        #pragma unroll
        for (int r = 0; r < 8; ++r) {
            int tok = mt * 16 + r + ((lane >> 4) << 3);
            int i = tok >> 3, j = tok & 7;
            int hd_ = ((hb * 8 + i) + SHIFT) & 255;
            int wd_ = ((wb * 8 + j) + SHIFT) & 255;
            xr[(((size_t)b * 65536) + hd_ * 256 + wd_) * CDIM + n] = acc[r] + bias;
        }
    }
}

// ---------------- LayerNorm helpers: one wave32 per token ----------------
__global__ __launch_bounds__(256) void k_ln1(const float* __restrict__ x,
                                             const float* __restrict__ xr,
                                             const float* __restrict__ g,
                                             const float* __restrict__ bt,
                                             float* __restrict__ out) {
    int wave = threadIdx.x >> 5, lane = threadIdx.x & 31;
    size_t tok = (size_t)blockIdx.x * 8 + wave;
    const float* row = xr + tok * CDIM;
    float v[6]; float s = 0.f;
    #pragma unroll
    for (int k = 0; k < 6; ++k) { v[k] = row[lane + 32 * k]; s += v[k]; }
    #pragma unroll
    for (int off = 16; off; off >>= 1) s += __shfl_xor(s, off, 32);
    float mu = s * (1.0f / CDIM);
    float q = 0.f;
    #pragma unroll
    for (int k = 0; k < 6; ++k) { float d = v[k] - mu; q += d * d; }
    #pragma unroll
    for (int off = 16; off; off >>= 1) q += __shfl_xor(q, off, 32);
    float inv = rsqrtf(q * (1.0f / CDIM) + 1e-5f);
    #pragma unroll
    for (int k = 0; k < 6; ++k) {
        int c = lane + 32 * k;
        out[tok * CDIM + c] = x[tok * CDIM + c] + (v[k] - mu) * inv * g[c] + bt[c];
    }
}

__global__ __launch_bounds__(256) void k_ln2(const float* __restrict__ mbuf,
                                             const float* __restrict__ g,
                                             const float* __restrict__ bt,
                                             float* __restrict__ out) {
    int wave = threadIdx.x >> 5, lane = threadIdx.x & 31;
    size_t tok = (size_t)blockIdx.x * 8 + wave;
    const float* row = mbuf + tok * CDIM;
    float v[6]; float s = 0.f;
    #pragma unroll
    for (int k = 0; k < 6; ++k) { v[k] = row[lane + 32 * k]; s += v[k]; }
    #pragma unroll
    for (int off = 16; off; off >>= 1) s += __shfl_xor(s, off, 32);
    float mu = s * (1.0f / CDIM);
    float q = 0.f;
    #pragma unroll
    for (int k = 0; k < 6; ++k) { float d = v[k] - mu; q += d * d; }
    #pragma unroll
    for (int off = 16; off; off >>= 1) q += __shfl_xor(q, off, 32);
    float inv = rsqrtf(q * (1.0f / CDIM) + 1e-5f);
    #pragma unroll
    for (int k = 0; k < 6; ++k) {
        int c = lane + 32 * k;
        out[tok * CDIM + c] = out[tok * CDIM + c] + (v[k] - mu) * inv * g[c] + bt[c];
    }
}

// ---------------- K5: fc1 + exact GELU (reads x1 fp32 from d_out) ----------------
__global__ __launch_bounds__(256) void k_fc1(const float* __restrict__ x1,
                                             const float* __restrict__ w,
                                             const float* __restrict__ bias,
                                             unsigned short* __restrict__ hbuf) {
    __shared__ alignas(16) unsigned short lA[64 * LSTRIDE];
    __shared__ alignas(16) unsigned short lB[64 * LSTRIDE];
    size_t m0 = (size_t)blockIdx.x * 64;
    int n0 = blockIdx.y * 64;
    for (int idx = threadIdx.x; idx < 64 * CDIM; idx += 256) {
        int t = idx / CDIM, c = idx % CDIM;
        lA[t * LSTRIDE + c] = f2bf(x1[(m0 + t) * CDIM + c]);
    }
    for (int idx = threadIdx.x; idx < 64 * CDIM; idx += 256) {
        int n = idx / CDIM, c = idx % CDIM;
        lB[n * LSTRIDE + c] = f2bf(w[(size_t)(n0 + n) * CDIM + c]);
    }
    __syncthreads();
    int wave = threadIdx.x >> 5, lane = threadIdx.x & 31;
    for (int t = 0; t < 2; ++t) {
        int tile = wave * 2 + t, mt = tile >> 2, nt = tile & 3;
        f32x8 acc = {0.f,0.f,0.f,0.f,0.f,0.f,0.f,0.f};
        #pragma unroll
        for (int kc = 0; kc < CDIM; kc += 32)
            acc = wmma_bf16(load_a_frag(lA, mt * 16, kc, LSTRIDE),
                            load_b_frag(lB, nt * 16, kc, LSTRIDE), acc);
        int n = n0 + nt * 16 + (lane & 15);
        float b = bias[n];
        #pragma unroll
        for (int r = 0; r < 8; ++r) {
            size_t row = m0 + mt * 16 + r + ((lane >> 4) << 3);
            float val = acc[r] + b;
            float ge  = 0.5f * val * (1.0f + erff(val * 0.70710678118654752f));
            hbuf[row * HIDDEN + n] = f2bf(ge);
        }
    }
}

// ---------------- K6: fc2 (K=768, chunked by 192; async A-tile, prefetch next W) ----------------
__global__ __launch_bounds__(256) void k_fc2(const unsigned short* __restrict__ hbuf,
                                             const float* __restrict__ w,
                                             const float* __restrict__ bias,
                                             float* __restrict__ mout) {
    __shared__ alignas(16) unsigned short lA[64 * LSTRIDE];
    __shared__ alignas(16) unsigned short lB[64 * LSTRIDE];
    size_t m0 = (size_t)blockIdx.x * 64;
    int n0 = blockIdx.y * 64;
    int wave = threadIdx.x >> 5, lane = threadIdx.x & 31;
    f32x8 acc[2] = {{0.f,0.f,0.f,0.f,0.f,0.f,0.f,0.f},
                    {0.f,0.f,0.f,0.f,0.f,0.f,0.f,0.f}};
    for (int ck = 0; ck < HIDDEN; ck += CDIM) {
        __syncthreads();
        // A tile: pure bf16 copy -> async load-to-LDS
        {
            const unsigned short* gbase = hbuf + m0 * HIDDEN + ck;
            #pragma unroll
            for (int it = 0; it < 6; ++it) {
                int chunk = threadIdx.x + 256 * it;    // 1536 x 16B chunks
                int row = (chunk * 8) / CDIM;
                int col = (chunk * 8) % CDIM;
                unsigned lds_off = (unsigned)(uintptr_t)&lA[row * LSTRIDE + col];
                unsigned goff    = (unsigned)((row * HIDDEN + col) * 2);
                async_copy_b128(lds_off, goff, gbase);
            }
        }
        for (int idx = threadIdx.x; idx < 64 * CDIM; idx += 256) {
            int n = idx / CDIM, c = idx % CDIM;
            lB[n * LSTRIDE + c] = f2bf(w[(size_t)(n0 + n) * HIDDEN + ck + c]);
        }
        // hint the next weight chunk into cache while we compute this one
        if (ck + CDIM < HIDDEN) {
            int prow = threadIdx.x & 63;
            int pseg = threadIdx.x >> 6;               // 4 probes per 768B row-chunk
            __builtin_prefetch(w + (size_t)(n0 + prow) * HIDDEN + ck + CDIM + pseg * 48, 0, 1);
        }
        wait_async0();
        __syncthreads();
        for (int t = 0; t < 2; ++t) {
            int tile = wave * 2 + t, mt = tile >> 2, nt = tile & 3;
            #pragma unroll
            for (int kc = 0; kc < CDIM; kc += 32)
                acc[t] = wmma_bf16(load_a_frag(lA, mt * 16, kc, LSTRIDE),
                                   load_b_frag(lB, nt * 16, kc, LSTRIDE), acc[t]);
        }
    }
    for (int t = 0; t < 2; ++t) {
        int tile = wave * 2 + t, mt = tile >> 2, nt = tile & 3;
        int n = n0 + nt * 16 + (lane & 15);
        float b = bias[n];
        #pragma unroll
        for (int r = 0; r < 8; ++r) {
            size_t row = m0 + mt * 16 + r + ((lane >> 4) << 3);
            mout[row * CDIM + n] = acc[t][r] + b;
        }
    }
}

// ---------------- host-side launch ----------------
extern "C" void kernel_launch(void* const* d_in, const int* in_sizes, int n_in,
                              void* d_out, int out_size, void* d_ws, size_t ws_size,
                              hipStream_t stream) {
    (void)in_sizes; (void)n_in; (void)out_size; (void)ws_size;
    const float* x       = (const float*)d_in[0];
    const float* mask    = (const float*)d_in[1];
    const int*   rel     = (const int*)  d_in[2];
    const float* qkv_w   = (const float*)d_in[3];
    const float* qkv_b   = (const float*)d_in[4];
    const float* proj_w  = (const float*)d_in[5];
    const float* proj_b  = (const float*)d_in[6];
    const float* rpb     = (const float*)d_in[7];
    const float* n1g     = (const float*)d_in[8];
    const float* n1b     = (const float*)d_in[9];
    const float* n2g     = (const float*)d_in[10];
    const float* n2b     = (const float*)d_in[11];
    const float* fc1_w   = (const float*)d_in[12];
    const float* fc1_b   = (const float*)d_in[13];
    const float* fc2_w   = (const float*)d_in[14];
    const float* fc2_b   = (const float*)d_in[15];
    float* out = (float*)d_out;

    char* ws = (char*)d_ws;
    // region layout (bytes):
    //   [0, 302M)           qkv bf16           (4096*3*6*64*32)
    //   [302M, 403M)        attn_out bf16      (4096*64*192)
    //   [403M region reuse] hbuf bf16 = [0, 403M)   (262144*768, after qkv/ao consumed)
    //   [403M, 604M)        xr fp32 / m fp32   (262144*192)
    unsigned short* qkv_ws = (unsigned short*)(ws);
    unsigned short* ao     = (unsigned short*)(ws + 301989888ull);
    float*          xr     = (float*)(ws + 402653184ull);
    unsigned short* hbuf   = (unsigned short*)(ws);        // reuse qkv+ao region
    float*          mbuf   = xr;                           // reuse xr region

    k_qkv <<<dim3(NWTOT, 9),  256, 0, stream>>>(x, qkv_w, qkv_b, qkv_ws);
    k_attn<<<dim3(NWTOT, NHEAD), 256, 0, stream>>>(qkv_ws, rpb, rel, mask, ao);
    k_proj<<<dim3(NWTOT, 3),  256, 0, stream>>>(ao, proj_w, proj_b, xr);
    k_ln1 <<<dim3(262144 / 8), 256, 0, stream>>>(x, xr, n1g, n1b, out);
    k_fc1 <<<dim3(NWTOT, 12), 256, 0, stream>>>(out, fc1_w, fc1_b, hbuf);
    k_fc2 <<<dim3(NWTOT, 3),  256, 0, stream>>>(hbuf, fc2_w, fc2_b, mbuf);
    k_ln2 <<<dim3(262144 / 8), 256, 0, stream>>>(mbuf, n2g, n2b, out);
}
